// DotProductAttention_75866302316691
// MI455X (gfx1250) — matrix operand order repaired and checked by reading, not compiled, not run on hardware
//
#include <hip/hip_runtime.h>
#include <cstddef>

typedef __attribute__((ext_vector_type(16))) _Float16 v16h;
typedef __attribute__((ext_vector_type(8)))  float    v8f;

namespace {
constexpr int kB = 4;
constexpr int kS = 4096;
constexpr int kD = 64;
constexpr int kQT = 16;                        // q rows per workgroup
constexpr int kNW = 8;                         // waves per workgroup
constexpr int kThreads = kNW * 32;             // 256
constexpr int kTilesPerWave = (kS / 16) / kNW; // 32 k-tiles per wave
constexpr int kKvPerWave = kTilesPerWave * 16; // 512 kv columns per wave
constexpr int kChunks = kTilesPerWave / 2;     // 16 chunks of 32 kv
}

__global__ __launch_bounds__(kThreads, 1)
void attn_fused(const float* __restrict__ Q, const float* __restrict__ V,
                float* __restrict__ ctx_out, float* __restrict__ attn_out)
{
  __shared__ _Float16 Qlds[kQT][kD];          // 2 KB   Q tile in f16
  __shared__ float    wred_m[kNW][kQT];       // cross-wave running max
  __shared__ float    wred_l[kNW][kQT];       // cross-wave running sum
  __shared__ float    rowmax_s[kQT];
  __shared__ float    rowinv_s[kQT];
  __shared__ _Float16 pscr[kNW][kQT][32];     // 8 KB   per-wave P chunk (A staging)
  __shared__ _Float16 vscr[kNW][32][kD];      // 32 KB  per-wave V chunk (B staging)
  __shared__ float    ctxred[kQT][kD];        // 4 KB   context reduction

  const int tid  = threadIdx.x;
  const int w    = tid >> 5;                  // wave id 0..7
  const int l    = tid & 31;                  // lane 0..31
  const int half = l >> 4;                    // 0 or 1 (16-lane half)
  const int lan  = l & 15;
  const int b    = blockIdx.y;
  const int q0   = blockIdx.x * kQT;

  const float* Qb = Q + ((size_t)b * kS + q0) * kD;
  const float* Vb = V + (size_t)b * kS * kD;

  for (int i = tid; i < kQT * kD; i += kThreads) {
    ((float*)ctxred)[i]  = 0.0f;
    ((_Float16*)Qlds)[i] = (_Float16)Qb[i];
  }
  __syncthreads();

  // ---- A operand (Q, 16x32 f16 per WMMA; two K-halves of D=64) ----
  // A order used here (contiguous k, same as B): lane(0..15)=row M; element e: K = half*16 + e (0..15)
  const int koffA = half * 16;
  v16h a0, a1;
#pragma unroll
  for (int e = 0; e < 8; ++e) {
    a0[e]     = Qlds[lan][koffA + e];
    a0[e + 8] = Qlds[lan][8 + koffA + e];
    a1[e]     = Qlds[lan][32 + koffA + e];
    a1[e + 8] = Qlds[lan][40 + koffA + e];
  }

  // ================= Pass A: online softmax statistics =================
  // B layout (32x16): lane holds N=lan (kv row in tile), K = half*16 + e (=d).
  const int doffB = half * 16;
  const int mrow  = half * 8;

  float m_r[8], l_r[8];
#pragma unroll
  for (int r = 0; r < 8; ++r) { m_r[r] = -3.402823466e38f; l_r[r] = 0.0f; }

  for (int t = 0; t < kTilesPerWave; ++t) {
    const int kv = w * kKvPerWave + t * 16 + lan;
    const float* vrow = Vb + (size_t)kv * kD;
    __builtin_prefetch(vrow + 16 * kD, 0, 1);
    v16h b0, b1;
#pragma unroll
    for (int e = 0; e < 16; ++e) {
      b0[e] = (_Float16)vrow[doffB + e];
      b1[e] = (_Float16)vrow[32 + doffB + e];
    }
    v8f c = {};
    c = __builtin_amdgcn_wmma_f32_16x16x32_f16(false, a0, false, b0, (short)0, c, false, false);
    c = __builtin_amdgcn_wmma_f32_16x16x32_f16(false, a1, false, b1, (short)0, c, false, false);
#pragma unroll
    for (int r = 0; r < 8; ++r) {              // online (m,l) update, scores discarded
      float s  = c[r];
      float mn = fmaxf(m_r[r], s);
      l_r[r]   = l_r[r] * __expf(m_r[r] - mn) + __expf(s - mn);
      m_r[r]   = mn;
    }
  }

  // cross-lane merge of (m,l) within each 16-lane half (wave32 shuffles)
#pragma unroll
  for (int off = 1; off <= 8; off <<= 1) {
#pragma unroll
    for (int r = 0; r < 8; ++r) {
      float mo = __shfl_xor(m_r[r], off, 32);
      float lo = __shfl_xor(l_r[r], off, 32);
      float mn = fmaxf(m_r[r], mo);
      l_r[r]   = l_r[r] * __expf(m_r[r] - mn) + lo * __expf(mo - mn);
      m_r[r]   = mn;
    }
  }
  if (lan == 0) {                              // lane 0 -> rows 0..7, lane 16 -> rows 8..15
#pragma unroll
    for (int r = 0; r < 8; ++r) { wred_m[w][mrow + r] = m_r[r]; wred_l[w][mrow + r] = l_r[r]; }
  }
  __syncthreads();
  if (tid < kQT) {                             // cross-wave merge
    float m = wred_m[0][tid], lsum = wred_l[0][tid];
#pragma unroll
    for (int ww = 1; ww < kNW; ++ww) {
      float mo = wred_m[ww][tid], lo = wred_l[ww][tid];
      float mn = fmaxf(m, mo);
      lsum = lsum * __expf(m - mn) + lo * __expf(mo - mn);
      m = mn;
    }
    rowmax_s[tid] = m;
    rowinv_s[tid] = 1.0f / lsum;
  }
  __syncthreads();

  float rm[8], rinv[8];
#pragma unroll
  for (int r = 0; r < 8; ++r) { rm[r] = rowmax_s[mrow + r]; rinv[r] = rowinv_s[mrow + r]; }

  // ========== Pass B: recompute scores, write attn, accumulate P@V ==========
  v8f cacc[4];
#pragma unroll
  for (int nt = 0; nt < 4; ++nt) cacc[nt] = (v8f){0,0,0,0,0,0,0,0};

  float* attnB = attn_out + ((size_t)b * kS + q0) * (size_t)kS;

  for (int cch = 0; cch < kChunks; ++cch) {
    const int kvbase = w * kKvPerWave + cch * 32;

    // stage V chunk rows -> f16 LDS (lane l owns row kvbase+l); single global read
    const float* vrow = Vb + (size_t)(kvbase + l) * kD;
#pragma unroll
    for (int d4 = 0; d4 < kD; d4 += 4) {
      float4 vv = *(const float4*)(vrow + d4);
      vscr[w][l][d4 + 0] = (_Float16)vv.x;
      vscr[w][l][d4 + 1] = (_Float16)vv.y;
      vscr[w][l][d4 + 2] = (_Float16)vv.z;
      vscr[w][l][d4 + 3] = (_Float16)vv.w;
    }
    asm volatile("s_wait_dscnt 0" ::: "memory");   // wave-private LDS RAW

    // recompute the chunk's two score tiles from vscr, normalize, emit attn + P
#pragma unroll
    for (int tt = 0; tt < 2; ++tt) {
      v16h b0, b1;                                 // B[k=d][n=kv]: contiguous along d
#pragma unroll
      for (int e = 0; e < 16; ++e) {
        b0[e] = vscr[w][tt * 16 + lan][doffB + e];
        b1[e] = vscr[w][tt * 16 + lan][32 + doffB + e];
      }
      v8f c = {};
      c = __builtin_amdgcn_wmma_f32_16x16x32_f16(false, a0, false, b0, (short)0, c, false, false);
      c = __builtin_amdgcn_wmma_f32_16x16x32_f16(false, a1, false, b1, (short)0, c, false, false);
      const int col = kvbase + tt * 16 + lan;
#pragma unroll
      for (int r = 0; r < 8; ++r) {
        float p = __expf(c[r] - rm[r]) * rinv[r];
        attnB[(size_t)(mrow + r) * kS + col] = p;          // the one mandatory write
        pscr[w][mrow + r][tt * 16 + lan] = (_Float16)p;    // A-staging for P@V
      }
    }
    asm volatile("s_wait_dscnt 0" ::: "memory");

    // A = P chunk (16x32 f16), same A layout as Q
    v16h ap;
#pragma unroll
    for (int e = 0; e < 8; ++e) {
      ap[e]     = pscr[w][lan][koffA + e];
      ap[e + 8] = pscr[w][lan][8 + koffA + e];
    }
    // B = V chunk: B[k=kv][n=d]; lane k = half*16 + e (column gather from vscr)
#pragma unroll
    for (int nt = 0; nt < 4; ++nt) {
      v16h bv;
#pragma unroll
      for (int e = 0; e < 16; ++e)
        bv[e] = vscr[w][half * 16 + e][nt * 16 + lan];
      cacc[nt] = __builtin_amdgcn_wmma_f32_16x16x32_f16(false, ap, false, bv,
                                                        (short)0, cacc[nt], false, false);
    }
  }

  // ---- cross-wave reduce via LDS float atomics, then store context ----
#pragma unroll
  for (int nt = 0; nt < 4; ++nt)
#pragma unroll
    for (int r = 0; r < 8; ++r)
      atomicAdd(&ctxred[mrow + r][nt * 16 + lan], cacc[nt][r]);
  __syncthreads();

  float* ctxB = ctx_out + ((size_t)b * kS + q0) * kD;
  for (int i = tid; i < kQT * kD; i += kThreads) ctxB[i] = ((float*)ctxred)[i];
}

extern "C" void kernel_launch(void* const* d_in, const int* in_sizes, int n_in,
                              void* d_out, int out_size, void* d_ws, size_t ws_size,
                              hipStream_t stream) {
  (void)in_sizes; (void)n_in; (void)out_size; (void)d_ws; (void)ws_size;
  const float* Q = (const float*)d_in[0];
  const float* V = (const float*)d_in[1];
  float* ctx  = (float*)d_out;                          // [B,S,D]
  float* attn = (float*)d_out + (size_t)kB * kS * kD;   // [B,S,S]
  dim3 grid(kS / kQT, kB);
  attn_fused<<<grid, kThreads, 0, stream>>>(Q, V, ctx, attn);
}
